// NSF_CL_70188355551501
// MI455X (gfx1250) — matrix operand order repaired and checked by reading, not compiled, not run on hardware
//
#include <hip/hip_runtime.h>

typedef _Float16 f16;
typedef __attribute__((ext_vector_type(16))) _Float16 v16h;
typedef __attribute__((ext_vector_type(8)))  float    v8f;

#define N_TOT   131072
#define HID     256
#define OD      736      // (3K-1)*32
#define ROWS    32       // rows per wave (two 16-row WMMA slabs)
#define ZSTR    40       // 32 + 8 f16 pad
#define HSTR    264      // 256 + 8 f16 pad
#define OSTR    736      // no pad needed (scalar spline reads)

// ws layout (f16 elements), per net: w1t [256][32] | w2t [256][256] | w3t [736][256]
#define W1T_OFF 0
#define W2T_OFF 8192
#define W3T_OFF 73728
#define NET_SZ  262144   // f16 elements per net

// ---------------- fast branchless transcendentals (raw TRANS ops) -----------
#define LOG2E 1.44269504088896340736f
#define LN2   0.69314718055994530942f

__device__ __forceinline__ float fast_exp(float x)  { return __builtin_amdgcn_exp2f(x * LOG2E); }
__device__ __forceinline__ float fast_log(float x)  { return __builtin_amdgcn_logf(x) * LN2; }
__device__ __forceinline__ float fast_rcp(float x)  { return __builtin_amdgcn_rcpf(x); }

// tanh(x) = 1 - 2/(e^{2x}+1); branchless, exact saturation (rcp(inf)=0)
__device__ __forceinline__ float fast_tanh(float x)
{
    float e = __builtin_amdgcn_exp2f(x * (2.0f * LOG2E));
    return 1.0f - 2.0f * fast_rcp(e + 1.0f);
}

// softplus(x) = max(x,0) + log(1 + e^{-|x|}); branchless
__device__ __forceinline__ float softplus_f(float x)
{
    return fmaxf(x, 0.0f) + fast_log(1.0f + fast_exp(-fabsf(x)));
}

// ---------------- prep: transpose + fp32->f16 all weights into workspace ----
__global__ __launch_bounds__(256) void nsf_prep(
    const float* __restrict__ w1a, const float* __restrict__ w2a, const float* __restrict__ w3a,
    const float* __restrict__ w1b, const float* __restrict__ w2b, const float* __restrict__ w3b,
    f16* __restrict__ wt)
{
    int idx = blockIdx.x * 256 + threadIdx.x;       // 0 .. 524287
    int net = idx >> 18;                            // 262144 per net
    int r   = idx & (NET_SZ - 1);
    const float* w1 = net ? w1b : w1a;
    const float* w2 = net ? w2b : w2a;
    const float* w3 = net ? w3b : w3a;
    float v;
    if (r < W2T_OFF) {                              // w1t: [n(0..255)][k(0..31)]
        int n = r >> 5, k = r & 31;
        v = w1[k * 256 + n];
    } else if (r < W3T_OFF) {                       // w2t: [n][k], 256x256
        int r2 = r - W2T_OFF;
        int n = r2 >> 8, k = r2 & 255;
        v = w2[k * 256 + n];
    } else {                                        // w3t: [n(0..735)][k(0..255)]
        int r3 = r - W3T_OFF;
        int n = r3 >> 8, k = r3 & 255;
        v = w3[k * 736 + n];
    }
    wt[idx] = (f16)v;
}

// ---------------- WMMA fragment loaders (ISA 7.12.2 16-bit layouts) ---------
// A (16x32, f16): lanes 0-15 -> rows, K in {kh*8..kh*8+7} U {16+kh*8..}, kh=lane>>4
__device__ __forceinline__ v16h ld_a(const f16* h, int stride, int k0, int lane)
{
    int r  = lane & 15;
    int kh = (lane >> 4) << 3;
    const f16* p = h + r * stride + k0 + kh;
    v16h a;
#pragma unroll
    for (int j = 0; j < 8; ++j) { a[j] = p[j]; a[j + 8] = p[j + 16]; }
    return a;
}

// B (32x16, f16), weights pre-transposed [Nout][Kin]: lane -> col N=lane&15,
// K in {kh*8..kh*8+7} U {16+kh*8..}, kh=lane>>4 (matches ld_a)
__device__ __forceinline__ v16h ld_b(const f16* wt, int kin, int n0, int k0, int lane)
{
    const f16* q = wt + (n0 + (lane & 15)) * kin + (k0 + ((lane >> 4) << 3));
    v16h b;
#pragma unroll
    for (int j = 0; j < 8; ++j) { b[j] = q[j]; b[j + 8] = q[j + 16]; }
    return b;
}

// -------- one MLP layer on TWO 16-row slabs: [32,KIN] x [KIN,NOUT] ----------
// A fragments register-resident for the whole layer; each B fragment feeds two
// independent WMMA chains (fills WMMA->VALU hazard slots, halves weight traffic)
template <int KIN, int NOUT, bool TANH>
__device__ __forceinline__ void mlp_layer(const f16* __restrict__ hin, int sin,
                                          const f16* __restrict__ wt,
                                          const float* __restrict__ bias,
                                          f16* __restrict__ hout, int sout, int lane)
{
    constexpr int KT = KIN / 32;
    const int col = lane & 15;
    const int rb  = (lane >> 4) << 3;   // C/D layout: M = v + 8*(lane>=16)

    v16h a0[KT], a1[KT];
#pragma unroll
    for (int kt = 0; kt < KT; ++kt) {
        a0[kt] = ld_a(hin,            sin, kt * 32, lane);   // rows 0..15
        a1[kt] = ld_a(hin + 16 * sin, sin, kt * 32, lane);   // rows 16..31
    }

#pragma unroll 2
    for (int n0 = 0; n0 < NOUT; n0 += 16) {
        float bv = bias[n0 + col];
        v8f acc0, acc1;
#pragma unroll
        for (int v = 0; v < 8; ++v) { acc0[v] = bv; acc1[v] = bv; }
#pragma unroll
        for (int kt = 0; kt < KT; ++kt) {
            v16h b = ld_b(wt, KIN, n0, kt * 32, lane);
            acc0 = __builtin_amdgcn_wmma_f32_16x16x32_f16(
                false, a0[kt], false, b, (short)0, acc0, false, false);
            acc1 = __builtin_amdgcn_wmma_f32_16x16x32_f16(
                false, a1[kt], false, b, (short)0, acc1, false, false);
        }
#pragma unroll
        for (int v = 0; v < 8; ++v) {
            float o0 = acc0[v], o1 = acc1[v];
            if (TANH) { o0 = fast_tanh(o0); o1 = fast_tanh(o1); }
            hout[(rb + v) * sout + n0 + col]        = (f16)o0;
            hout[(16 + rb + v) * sout + n0 + col]   = (f16)o1;
        }
    }
}

// ---------------- rational-quadratic spline (forward), reference math -------
__device__ __forceinline__ float2 rqs_eval(float xv, const f16* prm)
{
    const float left = -2.5f, right = 3.5f;   // tail_bound 3, offset 0.5
    const float MINB = 1e-3f;
    bool  inside = (xv >= left) && (xv <= right);
    float xc = fminf(fmaxf(xv, left), right);

    float wv[8], hv[8];
    // widths: coupling does 6*softmax(raw); rqs then softmaxes again
    {
        float t[8]; float mx = -3.0e38f;
#pragma unroll
        for (int j = 0; j < 8; ++j) { t[j] = (float)prm[j]; mx = fmaxf(mx, t[j]); }
        float s = 0.f;
#pragma unroll
        for (int j = 0; j < 8; ++j) { t[j] = fast_exp(t[j] - mx); s += t[j]; }
        float sc = 6.0f * fast_rcp(s); float mx2 = -3.0e38f;
#pragma unroll
        for (int j = 0; j < 8; ++j) { t[j] *= sc; mx2 = fmaxf(mx2, t[j]); }
        float s2 = 0.f;
#pragma unroll
        for (int j = 0; j < 8; ++j) { t[j] = fast_exp(t[j] - mx2); s2 += t[j]; }
        float inv = (1.0f - 8.0f * MINB) * fast_rcp(s2);
#pragma unroll
        for (int j = 0; j < 8; ++j) wv[j] = MINB + t[j] * inv;
    }
    // heights: identical pipeline on prm[8..15]
    {
        float t[8]; float mx = -3.0e38f;
#pragma unroll
        for (int j = 0; j < 8; ++j) { t[j] = (float)prm[8 + j]; mx = fmaxf(mx, t[j]); }
        float s = 0.f;
#pragma unroll
        for (int j = 0; j < 8; ++j) { t[j] = fast_exp(t[j] - mx); s += t[j]; }
        float sc = 6.0f * fast_rcp(s); float mx2 = -3.0e38f;
#pragma unroll
        for (int j = 0; j < 8; ++j) { t[j] *= sc; mx2 = fmaxf(mx2, t[j]); }
        float s2 = 0.f;
#pragma unroll
        for (int j = 0; j < 8; ++j) { t[j] = fast_exp(t[j] - mx2); s2 += t[j]; }
        float inv = (1.0f - 8.0f * MINB) * fast_rcp(s2);
#pragma unroll
        for (int j = 0; j < 8; ++j) hv[j] = MINB + t[j] * inv;
    }
    float cw[9], ch[9];
    cw[0] = left; ch[0] = left;
    float rw = 0.f, rh = 0.f;
#pragma unroll
    for (int j = 0; j < 8; ++j) {
        rw += wv[j]; cw[j + 1] = 6.0f * rw + left;
        rh += hv[j]; ch[j + 1] = 6.0f * rh + left;
    }
    cw[8] = right; ch[8] = right;
#pragma unroll
    for (int j = 0; j < 8; ++j) { wv[j] = cw[j + 1] - cw[j]; hv[j] = ch[j + 1] - ch[j]; }

    // derivatives: edge pad gives exactly MIN_D + (1-MIN_D) = 1.0
    float dv[9]; dv[0] = 1.0f; dv[8] = 1.0f;
#pragma unroll
    for (int j = 0; j < 7; ++j) {
        float rr = (float)prm[16 + j];                 // coupling: softplus, rqs: softplus again
        dv[j + 1] = MINB + softplus_f(softplus_f(rr));
    }

    // bin select via compare chain (== count(xc >= cumw[1..7]))
    float icw = cw[0], ibw = wv[0], ich = ch[0], ih = hv[0], d0 = dv[0], d1 = dv[1];
#pragma unroll
    for (int j = 1; j < 8; ++j) {
        bool c = xc >= cw[j];
        icw = c ? cw[j] : icw;  ibw = c ? wv[j] : ibw;
        ich = c ? ch[j] : ich;  ih  = c ? hv[j] : ih;
        d0  = c ? dv[j] : d0;   d1  = c ? dv[j + 1] : d1;
    }
    float rib  = fast_rcp(ibw);
    float idel = ih * rib;
    float th   = (xc - icw) * rib;
    float om   = 1.0f - th;
    float tom  = th * om;
    float numer = ih * (idel * th * th + d0 * tom);
    float den   = idel + (d0 + d1 - 2.0f * idel) * tom;
    float oin   = ich + numer * fast_rcp(den);
    float dnum  = idel * idel * (d1 * th * th + 2.0f * idel * tom + d0 * om * om);
    float lin   = fast_log(dnum) - 2.0f * fast_log(den);
    float2 res;
    res.x = inside ? oin : xv;
    res.y = inside ? lin : 0.0f;
    return res;
}

// ---------------- fused flow kernel: one wave per 32-row tile ---------------
// LDS lifetimes: sBig serves as H0 (layer1 out / layer2 in, first 16.9KB) and
// later as O (layer3 out / spline in, 47.1KB) -- disjoint in time.
// Total static LDS ~66.7KB -> 4 single-wave workgroups per 320KB WGP (all SIMDs).
__global__ __launch_bounds__(32) void nsf_main(
    const float* __restrict__ x,
    const f16*  __restrict__ wts,
    const float* __restrict__ b1a, const float* __restrict__ b2a, const float* __restrict__ b3a,
    const float* __restrict__ b1b, const float* __restrict__ b2b, const float* __restrict__ b3b,
    float* __restrict__ out)
{
    __shared__ __align__(16) f16   sBig[ROWS * OSTR];  // 47.1 KB (H0 | O, time-shared)
    __shared__ __align__(16) f16   sH1 [ROWS * HSTR];  // 16.9 KB
    __shared__ __align__(16) f16   sZ  [ROWS * ZSTR];  //  2.5 KB
    __shared__ float sLD[ROWS];

    f16* sH0 = sBig;   // stride HSTR view, layers 1-2
    f16* sO  = sBig;   // stride OSTR view, layer 3 + spline

    const int lane = threadIdx.x;
    const int row0 = blockIdx.x * ROWS;

    // stage lower half (cols 0..31) as f16 MLP input
#pragma unroll
    for (int i = 0; i < ROWS; ++i)
        sZ[i * ZSTR + lane] = (f16)x[(row0 + i) * 64 + lane];
    __syncthreads();

    // ---- coupling 1: net f1 on lower -> spline on upper
    mlp_layer<32,  HID, true >(sZ,  ZSTR, wts + W1T_OFF, b1a, sH0, HSTR, lane); __syncthreads();
    mlp_layer<HID, HID, true >(sH0, HSTR, wts + W2T_OFF, b2a, sH1, HSTR, lane); __syncthreads();
    mlp_layer<HID, OD,  false>(sH1, HSTR, wts + W3T_OFF, b3a, sO,  OSTR, lane); __syncthreads();

#pragma unroll 1
    for (int i = 0; i < ROWS; ++i) {
        float xv = x[(row0 + i) * 64 + 32 + lane];
        float2 r = rqs_eval(xv, &sO[i * OSTR + lane * 23]);
        out[(row0 + i) * 64 + 32 + lane] = r.x;     // final upper
        sZ[i * ZSTR + lane] = (f16)r.x;             // input to net f2
        float t = r.y;
#pragma unroll
        for (int off = 16; off > 0; off >>= 1) t += __shfl_xor(t, off, 32);
        if (lane == 0) sLD[i] = t;
    }
    __syncthreads();

    // ---- coupling 2: net f2 on upper' -> spline on lower
    mlp_layer<32,  HID, true >(sZ,  ZSTR, wts + NET_SZ + W1T_OFF, b1b, sH0, HSTR, lane); __syncthreads();
    mlp_layer<HID, HID, true >(sH0, HSTR, wts + NET_SZ + W2T_OFF, b2b, sH1, HSTR, lane); __syncthreads();
    mlp_layer<HID, OD,  false>(sH1, HSTR, wts + NET_SZ + W3T_OFF, b3b, sO,  OSTR, lane); __syncthreads();

#pragma unroll 1
    for (int i = 0; i < ROWS; ++i) {
        float xv = x[(row0 + i) * 64 + lane];
        float2 r = rqs_eval(xv, &sO[i * OSTR + lane * 23]);
        out[(row0 + i) * 64 + lane] = r.x;          // final lower
        float t = r.y;
#pragma unroll
        for (int off = 16; off > 0; off >>= 1) t += __shfl_xor(t, off, 32);
        if (lane == 0) out[(size_t)N_TOT * 64 + row0 + i] = sLD[i] + t;  // log_det
    }
}

// ---------------- host entry -----------------------------------------------
extern "C" void kernel_launch(void* const* d_in, const int* in_sizes, int n_in,
                              void* d_out, int out_size, void* d_ws, size_t ws_size,
                              hipStream_t stream)
{
    const float* x     = (const float*)d_in[0];
    const float* f1_w1 = (const float*)d_in[1];
    const float* f1_b1 = (const float*)d_in[2];
    const float* f1_w2 = (const float*)d_in[3];
    const float* f1_b2 = (const float*)d_in[4];
    const float* f1_w3 = (const float*)d_in[5];
    const float* f1_b3 = (const float*)d_in[6];
    const float* f2_w1 = (const float*)d_in[7];
    const float* f2_b1 = (const float*)d_in[8];
    const float* f2_w2 = (const float*)d_in[9];
    const float* f2_b2 = (const float*)d_in[10];
    const float* f2_w3 = (const float*)d_in[11];
    const float* f2_b3 = (const float*)d_in[12];

    f16* wt = (f16*)d_ws;   // 2 * 262144 f16 = 1 MiB of workspace

    nsf_prep<<<(2 * NET_SZ) / 256, 256, 0, stream>>>(
        f1_w1, f1_w2, f1_w3, f2_w1, f2_w2, f2_w3, wt);

    nsf_main<<<N_TOT / ROWS, 32, 0, stream>>>(
        x, wt, f1_b1, f1_b2, f1_b3, f2_b1, f2_b2, f2_b3, (float*)d_out);

    (void)in_sizes; (void)n_in; (void)out_size; (void)ws_size;
}
